// FusionGAT_31275951850258
// MI455X (gfx1250) — matrix operand order repaired and checked by reading, not compiled, not run on hardware
//
#include <hip/hip_runtime.h>
#include <hip/hip_bf16.h>
#include <math.h>

typedef __attribute__((ext_vector_type(16))) _Float16 v16h;
typedef __attribute__((ext_vector_type(8)))  _Float16 v8h;
typedef __attribute__((ext_vector_type(8)))  float    v8f;

#define F 128
#define CMAX 40

// ---------- order-preserving float<->uint encoding for atomicMax ----------
__device__ __forceinline__ unsigned fenc(float f) {
    unsigned u = __float_as_uint(f);
    return (u & 0x80000000u) ? ~u : (u | 0x80000000u);
}
__device__ __forceinline__ float fdec(unsigned u) {
    u = (u & 0x80000000u) ? (u & 0x7FFFFFFFu) : ~u;
    return __uint_as_float(u);
}

// ---------- tiny fill / convert kernels ----------
__global__ void fill_f32(float* p, float v, int n) {
    int i = blockIdx.x * blockDim.x + threadIdx.x;
    if (i < n) p[i] = v;
}
__global__ void fill_u32(unsigned* p, unsigned v, int n) {
    int i = blockIdx.x * blockDim.x + threadIdx.x;
    if (i < n) p[i] = v;
}
__global__ void f2h(_Float16* d, const float* s, int n) {
    int i = blockIdx.x * blockDim.x + threadIdx.x;
    if (i < n) d[i] = (_Float16)s[i];
}

// ---------- small dense prep: C[128x128] = A[128x128] * B[128x128] ----------
__global__ void mm128(float* __restrict__ C, const float* __restrict__ A,
                      const float* __restrict__ B) {
    int idx = blockIdx.x * blockDim.x + threadIdx.x;   // 16384 threads
    int i = idx >> 7, j = idx & 127;
    float s = 0.f;
    for (int k = 0; k < F; k++) s += A[i * F + k] * B[k * F + j];
    C[idx] = s;
}

// o[i] = sum_k A[i,k] v[k] + (b?b[i]:0)
__global__ void matvec128(float* o, const float* __restrict__ A,
                          const float* __restrict__ v, const float* b) {
    int i = threadIdx.x;
    float s = b ? b[i] : 0.f;
    for (int k = 0; k < F; k++) s += A[i * F + k] * v[k];
    o[i] = s;
}

// ---------- WMMA GEMM: out[n,128] = A[n,128] * W16[128,128]^T (+bias) ----------
// One wave computes a full 16x128 output strip: 8 accumulators, A-fragment
// loaded once per K-step and reused across 8 back-to-back WMMAs.
// AF32=true: A is float32, converted to f16 in-register (folds the cvt pass).
template <bool AF32>
__global__ __launch_bounds__(32) void wmma_gemm_k(
        float* __restrict__ out, const void* __restrict__ Ain,
        const _Float16* __restrict__ W, const float* __restrict__ bias, int n) {
    const int rowTile = blockIdx.x * 16;
    const int lane = threadIdx.x & 31;
    const int half = lane >> 4;            // 0 or 1
    const int mrow = lane & 15;
    const int abase = half * 16;           // A-frag K base (contiguous, matches B)
    const int bbase = half * 16;           // B-frag K base (contiguous, matches A)
    int rA = rowTile + mrow; if (rA >= n) rA = n - 1;

    const _Float16* Ah = (const _Float16*)Ain + (size_t)rA * F;
    const float*    Af = (const float*)Ain + (size_t)rA * F;
    const _Float16* Wb = W + (size_t)mrow * F;   // n-index within tile = lane&15

    v8f acc[8];
#pragma unroll
    for (int i = 0; i < 8; i++) acc[i] = (v8f){};

#pragma unroll
    for (int kk = 0; kk < F; kk += 32) {
        v16h af;
        if (AF32) {
            v8f g0 = *(const v8f*)(Af + kk + abase);        // K = base+0..7
            v8f g1 = *(const v8f*)(Af + kk + abase + 8);    // K = base+8..15
#pragma unroll
            for (int i = 0; i < 8; i++) { af[i] = (_Float16)g0[i]; af[i + 8] = (_Float16)g1[i]; }
        } else {
            v8h a0 = *(const v8h*)(Ah + kk + abase);
            v8h a1 = *(const v8h*)(Ah + kk + abase + 8);
#pragma unroll
            for (int i = 0; i < 8; i++) { af[i] = a0[i]; af[i + 8] = a1[i]; }
        }
#pragma unroll
        for (int ct = 0; ct < 8; ct++) {
            v16h bf = *(const v16h*)(Wb + (size_t)(ct * 16) * F + kk + bbase);
            acc[ct] = __builtin_amdgcn_wmma_f32_16x16x32_f16(
                          false, af, false, bf, (short)0, acc[ct], false, false);
        }
    }

    const int ncol = lane & 15;
    const int mofs = half * 8;
    float badd[8];
#pragma unroll
    for (int ct = 0; ct < 8; ct++) badd[ct] = bias ? bias[ct * 16 + ncol] : 0.f;

    if (rowTile + 16 <= n) {                      // wave-uniform fast path
        float* ob = out + (size_t)rowTile * F + ncol;
#pragma unroll
        for (int ct = 0; ct < 8; ct++)
#pragma unroll
            for (int r = 0; r < 8; r++)
                ob[(size_t)(r + mofs) * F + ct * 16] = acc[ct][r] + badd[ct];
    } else {
        for (int ct = 0; ct < 8; ct++)
            for (int r = 0; r < 8; r++) {
                int row = rowTile + r + mofs;
                if (row < n)
                    out[(size_t)row * F + ct * 16 + ncol] = acc[ct][r] + badd[ct];
            }
    }
}

// ---------- attention scalar dots: as[n]=H[n]·asrc, ad[n]=H[n]·adst ----------
__global__ void attdot(float* __restrict__ as_, float* __restrict__ ad_,
                       const float* __restrict__ H, const float* __restrict__ vs,
                       const float* __restrict__ vd, int n) {
    int node = blockIdx.x * (blockDim.x >> 5) + (threadIdx.x >> 5);
    int lane = threadIdx.x & 31;
    if (node >= n) return;
    const float* h = H + (size_t)node * F;
    float s1 = 0.f, s2 = 0.f;
    for (int f = lane; f < F; f += 32) { float x = h[f]; s1 += x * vs[f]; s2 += x * vd[f]; }
    for (int o = 16; o; o >>= 1) { s1 += __shfl_xor(s1, o, 32); s2 += __shfl_xor(s2, o, 32); }
    if (lane == 0) { as_[node] = s1; ad_[node] = s2; }
}

// ---------- edge softmax passes (self-loops appended as e in [E, E+n)) ----------
__device__ __forceinline__ void edge_sd(const int* ei, int E, int e, int& s, int& d) {
    if (e < E) { s = ei[e]; d = ei[E + e]; } else { s = e - E; d = e - E; }
}

__global__ void edge_max(const int* __restrict__ ei, int E, int n,
                         const float* __restrict__ as_, const float* __restrict__ ad_,
                         unsigned* __restrict__ m) {
    int e = blockIdx.x * blockDim.x + threadIdx.x;
    if (e >= E + n) return;
    int s, d; edge_sd(ei, E, e, s, d);
    float v = as_[s] + ad_[d];
    v = (v > 0.f) ? v : 0.2f * v;
    atomicMax(m + d, fenc(v));
}

__global__ void edge_exp(const int* __restrict__ ei, int E, int n,
                         const float* __restrict__ as_, const float* __restrict__ ad_,
                         const unsigned* __restrict__ m, float* __restrict__ ex,
                         float* __restrict__ ss) {
    int e = blockIdx.x * blockDim.x + threadIdx.x;
    if (e >= E + n) return;
    int s, d; edge_sd(ei, E, e, s, d);
    float v = as_[s] + ad_[d];
    v = (v > 0.f) ? v : 0.2f * v;
    float w = __expf(v - fdec(m[d]));
    ex[e] = w;
    atomicAdd(ss + d, w);
}

// one wave per edge: 128-wide feature scatter-add
__global__ void edge_agg(const int* __restrict__ ei, int E, int n,
                         const float* __restrict__ ex, const float* __restrict__ ss,
                         const float* __restrict__ H, float* __restrict__ AGG) {
    int e = blockIdx.x * (blockDim.x >> 5) + (threadIdx.x >> 5);
    int lane = threadIdx.x & 31;
    if (e >= E + n) return;
    int s, d; edge_sd(ei, E, e, s, d);
    float w = ex[e] / (ss[d] + 1e-16f);
    const float* hs = H + (size_t)s * F;
    float* od = AGG + (size_t)d * F;
    for (int f = lane; f < F; f += 32) atomicAdd(od + f, hs[f] * w);
}

// ---------- epilogues ----------
__global__ void epi_relu_h16(_Float16* __restrict__ h16, const float* __restrict__ AGG,
                             const float* __restrict__ b, int total) {
    int i = blockIdx.x * blockDim.x + threadIdx.x;
    if (i >= total) return;
    float v = AGG[i] + b[i & (F - 1)];
    v = v > 0.f ? v : 0.f;
    h16[i] = (_Float16)v;
}
__global__ void epi_bias(float* __restrict__ O, const float* __restrict__ b, int total) {
    int i = blockIdx.x * blockDim.x + threadIdx.x;
    if (i < total) O[i] += b[i & (F - 1)];
}

// ---------- final classifier + log_softmax ----------
__global__ void fc_logsoftmax(float* __restrict__ out, const float* __restrict__ ox,
                              const float* __restrict__ oy, const float* __restrict__ xr,
                              const float* __restrict__ fw, const float* __restrict__ fb,
                              int n, int C) {
    int node = blockIdx.x * blockDim.x + threadIdx.x;
    if (node >= n) return;
    const float* a = ox + (size_t)node * F;
    const float* bv = oy + (size_t)node * F;
    const float* c = xr + (size_t)node * F;
    float lg[CMAX];
    float mx = -3.4e38f;
    for (int cc = 0; cc < C; cc++) {
        const float* w = fw + (size_t)cc * (3 * F);
        float s = fb[cc];
        for (int k = 0; k < F; k++) s += a[k] * w[k];
        for (int k = 0; k < F; k++) s += bv[k] * w[F + k];
        for (int k = 0; k < F; k++) s += c[k] * w[2 * F + k];
        lg[cc] = s;
        mx = fmaxf(mx, s);
    }
    float se = 0.f;
    for (int cc = 0; cc < C; cc++) se += __expf(lg[cc] - mx);
    float lse = mx + __logf(se);
    for (int cc = 0; cc < C; cc++) out[(size_t)node * C + cc] = lg[cc] - lse;
}

// ---------- host-side helpers ----------
static char* carve(char*& p, size_t bytes) {
    char* r = p;
    p += (bytes + 255) & ~(size_t)255;
    return r;
}

static void run_edges(hipStream_t st, const int* ei, int E, int n,
                      const float* Hpre, float* as_, float* ad_,
                      const float* asrc, const float* adst,
                      unsigned* mbuf, float* sbuf, float* exbuf, float* AGG) {
    int tot = E + n;
    attdot<<<(n + 7) / 8, 256, 0, st>>>(as_, ad_, Hpre, asrc, adst, n);
    fill_u32<<<(n + 255) / 256, 256, 0, st>>>(mbuf, 0u, n);   // 0 = minimal encoded value
    fill_f32<<<(n + 255) / 256, 256, 0, st>>>(sbuf, 0.f, n);
    fill_f32<<<(n * F + 255) / 256, 256, 0, st>>>(AGG, 0.f, n * F);
    edge_max<<<(tot + 255) / 256, 256, 0, st>>>(ei, E, n, as_, ad_, mbuf);
    edge_exp<<<(tot + 255) / 256, 256, 0, st>>>(ei, E, n, as_, ad_, mbuf, exbuf, sbuf);
    edge_agg<<<(tot + 7) / 8, 256, 0, st>>>(ei, E, n, exbuf, sbuf, Hpre, AGG);
}

extern "C" void kernel_launch(void* const* d_in, const int* in_sizes, int n_in,
                              void* d_out, int out_size, void* d_ws, size_t ws_size,
                              hipStream_t stream) {
    const int n = in_sizes[0] / F;        // 50000
    const int E = in_sizes[2] / 2;        // 800000
    const int C = in_sizes[25];           // 40

    const float* x   = (const float*)d_in[0];
    const float* y   = (const float*)d_in[1];
    const int*   eix = (const int*)d_in[2];
    const int*   eiy = (const int*)d_in[3];
    const float* a1iw = (const float*)d_in[4];
    const float* a1ib = (const float*)d_in[5];
    const float* a1ow = (const float*)d_in[6];
    const float* a1ob = (const float*)d_in[7];
    const float* a2iw = (const float*)d_in[8];
    const float* a2ib = (const float*)d_in[9];
    const float* a2ow = (const float*)d_in[10];
    const float* a2ob = (const float*)d_in[11];
    const float* g1w  = (const float*)d_in[12];
    const float* g1as = (const float*)d_in[13];
    const float* g1ad = (const float*)d_in[14];
    const float* g1b  = (const float*)d_in[15];
    const float* g2xw = (const float*)d_in[16];
    const float* g2xas = (const float*)d_in[17];
    const float* g2xad = (const float*)d_in[18];
    const float* g2xb  = (const float*)d_in[19];
    const float* g2yw = (const float*)d_in[20];
    const float* g2yas = (const float*)d_in[21];
    const float* g2yad = (const float*)d_in[22];
    const float* g2yb  = (const float*)d_in[23];
    const float* fcw  = (const float*)d_in[24];
    const float* fcb  = (const float*)d_in[25];
    float* out = (float*)d_out;

    // ---- workspace carve ----
    char* p = (char*)d_ws;
    float*     T128   = (float*)carve(p, F * F * 4);
    float*     Mf     = (float*)carve(p, F * F * 4);
    _Float16*  MX16   = (_Float16*)carve(p, F * F * 2);
    _Float16*  MY16   = (_Float16*)carve(p, F * F * 2);
    _Float16*  W2X16  = (_Float16*)carve(p, F * F * 2);
    _Float16*  W2Y16  = (_Float16*)carve(p, F * F * 2);
    float*     cv     = (float*)carve(p, F * 4);
    float*     ceffx  = (float*)carve(p, F * 4);
    float*     ceffy  = (float*)carve(p, F * 4);
    _Float16*  IN16   = (_Float16*)carve(p, (size_t)n * F * 2);
    float*     Hpre   = (float*)carve(p, (size_t)n * F * 4);
    float*     AGG    = (float*)carve(p, (size_t)n * F * 4);
    float*     OX     = (float*)carve(p, (size_t)n * F * 4);
    float*     OY     = (float*)carve(p, (size_t)n * F * 4);
    float*     as_    = (float*)carve(p, (size_t)n * 4);
    float*     ad_    = (float*)carve(p, (size_t)n * 4);
    unsigned*  mbuf   = (unsigned*)carve(p, (size_t)n * 4);
    float*     sbuf   = (float*)carve(p, (size_t)n * 4);
    float*     exbuf  = (float*)carve(p, (size_t)(E + n) * 4);
    (void)ws_size; (void)n_in; (void)out_size;

    const int nF = n * F;
    const int gemmBlocks = (n + 15) / 16;

    // ---- fused weight prep: Meff = W1 * (Wo * Wv), ceff = W1 * (Wo*bv + bo) ----
    mm128<<<64, 256, 0, stream>>>(T128, a1ow, a1iw + 2 * F * F);
    mm128<<<64, 256, 0, stream>>>(Mf, g1w, T128);
    f2h<<<64, 256, 0, stream>>>(MX16, Mf, F * F);
    matvec128<<<1, 128, 0, stream>>>(cv, a1ow, a1ib + 2 * F, a1ob);
    matvec128<<<1, 128, 0, stream>>>(ceffx, g1w, cv, nullptr);

    mm128<<<64, 256, 0, stream>>>(T128, a2ow, a2iw + 2 * F * F);
    mm128<<<64, 256, 0, stream>>>(Mf, g1w, T128);
    f2h<<<64, 256, 0, stream>>>(MY16, Mf, F * F);
    matvec128<<<1, 128, 0, stream>>>(cv, a2ow, a2ib + 2 * F, a2ob);
    matvec128<<<1, 128, 0, stream>>>(ceffy, g1w, cv, nullptr);

    f2h<<<64, 256, 0, stream>>>(W2X16, g2xw, F * F);
    f2h<<<64, 256, 0, stream>>>(W2Y16, g2yw, F * F);

    // ---- x path ----
    wmma_gemm_k<true><<<gemmBlocks, 32, 0, stream>>>(Hpre, x, MX16, ceffx, n);
    run_edges(stream, eix, E, n, Hpre, as_, ad_, g1as, g1ad, mbuf, sbuf, exbuf, AGG);
    epi_relu_h16<<<(nF + 255) / 256, 256, 0, stream>>>(IN16, AGG, g1b, nF);
    wmma_gemm_k<false><<<gemmBlocks, 32, 0, stream>>>(Hpre, IN16, W2X16, nullptr, n);
    run_edges(stream, eix, E, n, Hpre, as_, ad_, g2xas, g2xad, mbuf, sbuf, exbuf, OX);
    epi_bias<<<(nF + 255) / 256, 256, 0, stream>>>(OX, g2xb, nF);

    // ---- y path ----
    wmma_gemm_k<true><<<gemmBlocks, 32, 0, stream>>>(Hpre, y, MY16, ceffy, n);
    run_edges(stream, eiy, E, n, Hpre, as_, ad_, g1as, g1ad, mbuf, sbuf, exbuf, AGG);
    epi_relu_h16<<<(nF + 255) / 256, 256, 0, stream>>>(IN16, AGG, g1b, nF);
    wmma_gemm_k<false><<<gemmBlocks, 32, 0, stream>>>(Hpre, IN16, W2Y16, nullptr, n);
    run_edges(stream, eiy, E, n, Hpre, as_, ad_, g2yas, g2yad, mbuf, sbuf, exbuf, OY);
    epi_bias<<<(nF + 255) / 256, 256, 0, stream>>>(OY, g2yb, nF);

    // ---- fusion + fc + log_softmax ----
    fc_logsoftmax<<<(n + 255) / 256, 256, 0, stream>>>(out, OX, OY, x, fcw, fcb, n, C);
}
